// Nucleus_60576218743078
// MI455X (gfx1250) — compile-verified, matrix operand order checked by reading, not run on hardware
//
#include <hip/hip_runtime.h>

// ---------------- model constants ----------------
constexpr int V_ = 50258, D_ = 1024, NHEAD_ = 8, HD_ = 128, NHID_ = 2048;
constexpr int NUIDS_ = 2000, B_ = 4, S_ = 256, TOPK_ = 20;

typedef __attribute__((ext_vector_type(16))) __bf16  v16bf;
typedef __attribute__((ext_vector_type(8)))  __bf16  bf8_t;
typedef __attribute__((ext_vector_type(8)))  float   v8f;
typedef __attribute__((ext_vector_type(8)))  float   f8_t;
typedef __attribute__((ext_vector_type(4)))  float   f4_t;
typedef __attribute__((ext_vector_type(8)))  unsigned short us8_t;
typedef __attribute__((ext_vector_type(4)))  unsigned int   u32x4;
typedef __attribute__((ext_vector_type(4)))  int            i32x4;
typedef __attribute__((ext_vector_type(8)))  int            i32x8;

// ---------------- helpers ----------------
__device__ __forceinline__ unsigned short bf_bits(float f) {
    union { __bf16 b; unsigned short s; } o;
    o.b = (__bf16)f;
    return o.s;
}

// pack 8 fp32 -> 8 bf16; msk (0.0/1.0) applied at fp32 (branchless, no sub-word ops)
__device__ __forceinline__ us8_t pack8(f4_t a, f4_t b, float msk) {
    f8_t f;
    #pragma unroll
    for (int j = 0; j < 4; ++j) { f[j] = a[j] * msk; f[4 + j] = b[j] * msk; }
    union { bf8_t b; us8_t u; } o;
    o.b = __builtin_convertvector(f, bf8_t);   // packed v_cvt_pk_bf16_f32 x4
    return o.u;
}

// assemble a 16-bf16 fragment from two 16-byte LDS vectors (K-chunks hi8, 16+hi8)
__device__ __forceinline__ v16bf frag_from(const us8_t* rowp, int hv) {
    union { us8_t v[2]; v16bf f; } o;
    o.v[0] = rowp[hv];
    o.v[1] = rowp[2 + hv];
    return o.f;
}

// ---------------- generic strided/batched WMMA GEMM ----------------
// C[m,n] = alpha * sum_k A[m,k] * (transB ? B[n,k] : B[k,n]) + bias[n], opt ReLU
// 128 threads = 4 waves; WG tile 64x64; each wave a 32x32 sub-tile (4 WMMAs/k-step)
__launch_bounds__(128)
__global__ void gemm_wmma(const float* __restrict__ A, long lda, long sAb, long sAh,
                          const float* __restrict__ Bm, long ldb, long sBb, long sBh,
                          const float* __restrict__ bias,
                          float* __restrict__ C, long ldc, long sCb, long sCh,
                          int M, int N, int K, int HZ,
                          float alpha, int transB, int relu)
{
    __shared__ us8_t Asb8[64][5];   // 64 rows x 32 k (bf16), 80B row stride
    __shared__ us8_t Bsb8[64][5];   // [n][k]

    int z  = blockIdx.z;
    int zb = z / HZ, zh = z % HZ;
    A  += (long)zb * sAb + (long)zh * sAh;
    Bm += (long)zb * sBb + (long)zh * sBh;
    C  += (long)zb * sCb + (long)zh * sCh;

    const int tid  = threadIdx.x;
    const int wave = tid >> 5;
    const int lane = tid & 31;
    const int mbase = blockIdx.y * 64;
    const int nbase = blockIdx.x * 64;
    const int tm = (wave >> 1) * 32, tn = (wave & 1) * 32;

    v8f acc[2][2] = {};

    const int srow  = tid >> 1;        // 0..63
    const int shalf = tid & 1;         // 0/1 -> k cols 0..15 / 16..31
    const int scol0 = shalf * 16;

    const int mn  = lane & 15;
    const int hv  = lane >> 4;         // 0/1 (K-chunk select)

    // row clamps + fp32 masks (branchless staging)
    const int   gmA  = mbase + srow;
    const long  gmAc = gmA < M ? gmA : (M - 1);
    const float mskA = gmA < M ? 1.f : 0.f;
    const int   gnB  = nbase + srow;
    const long  gnBc = gnB < N ? gnB : (N - 1);
    const float mskB = gnB < N ? 1.f : 0.f;

    for (int kb = 0; kb < K; kb += 32) {
        // ---- stage A tile (64 x 32) ----
        {
            const f4_t* ap = (const f4_t*)(A + gmAc * lda + kb + scol0);
            __builtin_prefetch(ap + 8, 0, 1);           // next K tile
            f4_t q0 = ap[0], q1 = ap[1], q2 = ap[2], q3 = ap[3];
            Asb8[srow][shalf * 2 + 0] = pack8(q0, q1, mskA);
            Asb8[srow][shalf * 2 + 1] = pack8(q2, q3, mskA);
        }
        // ---- stage B tile: Bsb[n][k] ----
        if (transB) {
            const f4_t* bp = (const f4_t*)(Bm + gnBc * ldb + kb + scol0);
            __builtin_prefetch(bp + 8, 0, 1);
            f4_t q0 = bp[0], q1 = bp[1], q2 = bp[2], q3 = bp[3];
            Bsb8[srow][shalf * 2 + 0] = pack8(q0, q1, mskB);
            Bsb8[srow][shalf * 2 + 1] = pack8(q2, q3, mskB);
        } else {
            // B is [K,N]; used only for fully in-range shapes (att @ V)
            int krow = tid >> 2;               // 0..31
            int c0   = (tid & 3) * 16;         // 0..48
            const f4_t* bp = (const f4_t*)(Bm + (long)(kb + krow) * ldb + nbase + c0);
            f4_t q[4] = { bp[0], bp[1], bp[2], bp[3] };
            unsigned short* bs = (unsigned short*)&Bsb8[0][0];   // row stride 40
            #pragma unroll
            for (int j = 0; j < 16; ++j) {
                float v = q[j >> 2][j & 3];
                bs[(c0 + j) * 40 + krow] = bf_bits(v);
            }
        }
        __syncthreads();

        v16bf a0 = frag_from(&Asb8[tm + mn][0], hv);
        v16bf a1 = frag_from(&Asb8[tm + 16 + mn][0], hv);
        v16bf b0 = frag_from(&Bsb8[tn + mn][0], hv);
        v16bf b1 = frag_from(&Bsb8[tn + 16 + mn][0], hv);
        acc[0][0] = __builtin_amdgcn_wmma_f32_16x16x32_bf16(false, a0, false, b0, (short)0, acc[0][0], false, false);
        acc[0][1] = __builtin_amdgcn_wmma_f32_16x16x32_bf16(false, a0, false, b1, (short)0, acc[0][1], false, false);
        acc[1][0] = __builtin_amdgcn_wmma_f32_16x16x32_bf16(false, a1, false, b0, (short)0, acc[1][0], false, false);
        acc[1][1] = __builtin_amdgcn_wmma_f32_16x16x32_bf16(false, a1, false, b1, (short)0, acc[1][1], false, false);
        __syncthreads();
    }

    // ---- epilogue ----
    const int hi8r = hv * 8;
    #pragma unroll
    for (int sn = 0; sn < 2; ++sn) {
        int gn = nbase + tn + sn * 16 + mn;
        if (gn < N) {
            float bv = bias ? bias[gn] : 0.f;
            #pragma unroll
            for (int sm = 0; sm < 2; ++sm) {
                #pragma unroll
                for (int i = 0; i < 8; ++i) {
                    int gm = mbase + tm + sm * 16 + i + hi8r;
                    if (gm < M) {
                        float v = alpha * acc[sm][sn][i] + bv;
                        if (relu) v = v > 0.f ? v : 0.f;
                        C[(long)gm * ldc + gn] = v;
                    }
                }
            }
        }
    }
}

// ---------------- softmax over rows of length 256 (one wave per row) -------
__launch_bounds__(256)
__global__ void softmax_rows(float* __restrict__ att)
{
    long wrow = (long)blockIdx.x * 8 + (threadIdx.x >> 5);
    int lane = threadIdx.x & 31;
    float* row = att + wrow * S_;
    float vals[8];
    float mx = -__builtin_inff();
    #pragma unroll
    for (int i = 0; i < 8; ++i) { vals[i] = row[lane + i * 32]; mx = fmaxf(mx, vals[i]); }
    #pragma unroll
    for (int off = 16; off > 0; off >>= 1) mx = fmaxf(mx, __shfl_xor(mx, off, 32));
    float s = 0.f;
    #pragma unroll
    for (int i = 0; i < 8; ++i) { vals[i] = __expf(vals[i] - mx); s += vals[i]; }
    #pragma unroll
    for (int off = 16; off > 0; off >>= 1) s += __shfl_xor(s, off, 32);
    float inv = 1.f / s;
    #pragma unroll
    for (int i = 0; i < 8; ++i) row[lane + i * 32] = vals[i] * inv;
}

// ---------------- fused residual-add + LayerNorm over D=1024 --------------
__launch_bounds__(256)
__global__ void add_ln(const float* __restrict__ a, const float* __restrict__ b,
                       const float* __restrict__ g, const float* __restrict__ be,
                       float* __restrict__ out)
{
    __shared__ float ssum[8], ssum2[8];
    long row = blockIdx.x;
    int t = threadIdx.x;
    const float* ar = a + row * D_;
    const float* br = b + row * D_;
    float x[4]; float s = 0.f, s2 = 0.f;
    #pragma unroll
    for (int i = 0; i < 4; ++i) {
        float v = ar[t + i * 256] + br[t + i * 256];
        x[i] = v; s += v; s2 += v * v;
    }
    #pragma unroll
    for (int off = 16; off > 0; off >>= 1) { s += __shfl_xor(s, off, 32); s2 += __shfl_xor(s2, off, 32); }
    if ((t & 31) == 0) { ssum[t >> 5] = s; ssum2[t >> 5] = s2; }
    __syncthreads();
    if (t == 0) {
        float S = 0.f, S2 = 0.f;
        for (int i = 0; i < 8; ++i) { S += ssum[i]; S2 += ssum2[i]; }
        ssum[0] = S / (float)D_;
        ssum2[0] = S2 / (float)D_;
    }
    __syncthreads();
    float mean = ssum[0];
    float var  = ssum2[0] - mean * mean;
    float rstd = rsqrtf(var + 1e-5f);
    float* orow = out + row * D_;
    #pragma unroll
    for (int i = 0; i < 4; ++i) {
        int c = t + i * 256;
        orow[c] = (x[i] - mean) * rstd * g[c] + be[c];
    }
}

// ---------------- embedding gather ----------------
__launch_bounds__(256)
__global__ void gather_emb(const int* __restrict__ ids, const float* __restrict__ emb,
                           float* __restrict__ out)
{
    long row = blockIdx.x;
    long id  = ids[row];
    const float* src = emb + id * D_;
    float* dst = out + row * D_;
    for (int i = threadIdx.x; i < D_; i += 256) dst[i] = src[i];
}

// ---------------- gate mean/std/noise + top-k selection -------------------
__launch_bounds__(256)
__global__ void gate_topk(const float* __restrict__ rw, const float* __restrict__ noise,
                          const int* __restrict__ topk, float* __restrict__ out)
{
    __shared__ float full[NUIDS_];
    __shared__ float rbuf[256];
    __shared__ int   ribuf[256];
    int t = threadIdx.x;
    float s = 0.f, s2 = 0.f;
    for (int u = t; u < NUIDS_; u += 256) {
        float bw = 0.25f * (rw[u] + rw[NUIDS_ + u] + rw[2 * NUIDS_ + u] + rw[3 * NUIDS_ + u]);
        full[u] = bw; s += bw; s2 += bw * bw;
    }
    rbuf[t] = s; __syncthreads();
    for (int off = 128; off > 0; off >>= 1) { if (t < off) rbuf[t] += rbuf[t + off]; __syncthreads(); }
    float mean = rbuf[0] / (float)NUIDS_;
    __syncthreads();
    rbuf[t] = s2; __syncthreads();
    for (int off = 128; off > 0; off >>= 1) { if (t < off) rbuf[t] += rbuf[t + off]; __syncthreads(); }
    float var = rbuf[0] / (float)NUIDS_ - mean * mean;
    float sd  = sqrtf(fmaxf(var, 0.f));
    __syncthreads();
    for (int u = t; u < NUIDS_; u += 256) full[u] += noise[u] * sd;
    __syncthreads();
    int K = topk[0];
    if (K > TOPK_) K = TOPK_;
    for (int k = 0; k < K; ++k) {
        float best = -__builtin_inff(); int bi = NUIDS_;
        for (int u = t; u < NUIDS_; u += 256) {
            float v = full[u];
            if (v > best) { best = v; bi = u; }
        }
        rbuf[t] = best; ribuf[t] = bi; __syncthreads();
        for (int off = 128; off > 0; off >>= 1) {
            if (t < off) {
                if (rbuf[t + off] > rbuf[t] ||
                    (rbuf[t + off] == rbuf[t] && ribuf[t + off] < ribuf[t])) {
                    rbuf[t] = rbuf[t + off]; ribuf[t] = ribuf[t + off];
                }
            }
            __syncthreads();
        }
        if (t == 0) {
            out[1 + k]         = rbuf[0];
            out[1 + TOPK_ + k] = (float)ribuf[0];
            full[ribuf[0]]     = -__builtin_inff();
        }
        __syncthreads();
    }
}

// ---------------- streaming decode + online log-softmax CE ----------------
// 64 WGs x 16 rows. A (16x1024) cached once in LDS as bf16; each wave sweeps
// the vocab in 32-col chunks (2 WMMAs / k-step). Per-lane online softmax over
// the lane's own columns; a single butterfly merge at the end.
__launch_bounds__(128)
__global__ void ce_kernel(const float* __restrict__ X, const float* __restrict__ W,
                          const int* __restrict__ tok, float* __restrict__ rowloss)
{
    __shared__ us8_t Asb8[16][129];               // 16 rows x 1024 k bf16
    __shared__ float Mm[4][2][8], Msum[4][2][8], Mlbl[4][2][8];

    const int tid  = threadIdx.x;
    const int wave = tid >> 5;
    const int lane = tid & 31;
    const int mn   = lane & 15;
    const int hv   = lane >> 4;
    const int hi8  = hv * 8;
    const int mbase = blockIdx.x * 16;

    for (int idx = tid; idx < 16 * 128; idx += 128) {
        int r = idx >> 7, vec = idx & 127;
        const f4_t* xp = (const f4_t*)(X + (long)(mbase + r) * D_ + vec * 8);
        Asb8[r][vec] = pack8(xp[0], xp[1], 1.f);
    }
    __syncthreads();

    float m_run[8], s_run[8], lbl[8];
    int lab[8];
    #pragma unroll
    for (int i = 0; i < 8; ++i) {
        m_run[i] = -__builtin_inff(); s_run[i] = 0.f; lbl[i] = 0.f;
        int r = mbase + i + hi8;
        lab[i] = ((r & (S_ - 1)) != S_ - 1) ? tok[r + 1] : -1;
    }

    for (int nb = 0; nb < V_; nb += 128) {
        int ncol = nb + wave * 32;
        int g0 = ncol + mn, g1 = ncol + 16 + mn;
        bool ok0 = g0 < V_, ok1 = g1 < V_;
        const float* bp0 = W + (long)(ok0 ? g0 : 0) * D_;
        const float* bp1 = W + (long)(ok1 ? g1 : 0) * D_;

        v8f acc0 = {}, acc1 = {};
        for (int kb = 0; kb < D_; kb += 32) {
            const us8_t* ar = &Asb8[mn][kb >> 3];
            union { us8_t v[2]; v16bf f; } ua;
            ua.v[0] = ar[hv]; ua.v[1] = ar[2 + hv];

            const f4_t* p00 = (const f4_t*)(bp0 + kb + hi8);
            const f4_t* p01 = (const f4_t*)(bp0 + kb + 16 + hi8);
            const f4_t* p10 = (const f4_t*)(bp1 + kb + hi8);
            const f4_t* p11 = (const f4_t*)(bp1 + kb + 16 + hi8);
            union { us8_t v[2]; v16bf f; } ub0, ub1;
            ub0.v[0] = pack8(p00[0], p00[1], 1.f);
            ub0.v[1] = pack8(p01[0], p01[1], 1.f);
            ub1.v[0] = pack8(p10[0], p10[1], 1.f);
            ub1.v[1] = pack8(p11[0], p11[1], 1.f);

            acc0 = __builtin_amdgcn_wmma_f32_16x16x32_bf16(false, ua.f, false, ub0.f, (short)0, acc0, false, false);
            acc1 = __builtin_amdgcn_wmma_f32_16x16x32_bf16(false, ua.f, false, ub1.f, (short)0, acc1, false, false);
        }
        // per-lane online softmax update over this lane's two columns
        #pragma unroll
        for (int i = 0; i < 8; ++i) {
            float v0 = ok0 ? acc0[i] : -__builtin_inff();
            float nm0 = fmaxf(m_run[i], v0);
            s_run[i] = s_run[i] * __expf(m_run[i] - nm0) + __expf(v0 - nm0);
            m_run[i] = nm0;
            lbl[i] += (g0 == lab[i]) ? acc0[i] : 0.f;

            float v1 = ok1 ? acc1[i] : -__builtin_inff();
            float nm1 = fmaxf(m_run[i], v1);
            s_run[i] = s_run[i] * __expf(m_run[i] - nm1) + __expf(v1 - nm1);
            m_run[i] = nm1;
            lbl[i] += (g1 == lab[i]) ? acc1[i] : 0.f;
        }
    }

    // butterfly merge across the 16 lanes of each half
    #pragma unroll
    for (int i = 0; i < 8; ++i) {
        #pragma unroll
        for (int off = 1; off < 16; off <<= 1) {
            float mo = __shfl_xor(m_run[i], off, 32);
            float so = __shfl_xor(s_run[i], off, 32);
            float lo = __shfl_xor(lbl[i],  off, 32);
            float nm = fmaxf(m_run[i], mo);
            s_run[i] = s_run[i] * __expf(m_run[i] - nm) + so * __expf(mo - nm);
            m_run[i] = nm;
            lbl[i] += lo;
        }
    }
    if (mn == 0) {
        #pragma unroll
        for (int i = 0; i < 8; ++i) {
            Mm[wave][hv][i] = m_run[i]; Msum[wave][hv][i] = s_run[i]; Mlbl[wave][hv][i] = lbl[i];
        }
    }
    __syncthreads();
    if (wave == 0 && lane < 16) {
        int i = lane & 7, hi = lane >> 3;
        float M = -__builtin_inff(), S = 0.f, L = 0.f;
        for (int w = 0; w < 4; ++w) M = fmaxf(M, Mm[w][hi][i]);
        for (int w = 0; w < 4; ++w) { S += Msum[w][hi][i] * __expf(Mm[w][hi][i] - M); L += Mlbl[w][hi][i]; }
        int r = mbase + lane;
        float loss = ((r & (S_ - 1)) != S_ - 1) ? (M + __logf(S) - L) : 0.f;
        rowloss[r] = loss;
    }
}

__launch_bounds__(256)
__global__ void loss_reduce(const float* __restrict__ rowloss, float* __restrict__ out)
{
    __shared__ float sb[256];
    float s = 0.f;
    for (int i = threadIdx.x; i < B_ * S_; i += 256) s += rowloss[i];
    sb[threadIdx.x] = s; __syncthreads();
    for (int off = 128; off > 0; off >>= 1) {
        if (threadIdx.x < off) sb[threadIdx.x] += sb[threadIdx.x + off];
        __syncthreads();
    }
    if (threadIdx.x == 0) out[0] = sb[0] / (float)(B_ * (S_ - 1));
}

// ---------------- CDNA5 probe: TDM NULL descriptor + cluster ops ----------
#if defined(__has_builtin)
#if __has_builtin(__builtin_amdgcn_tensor_load_to_lds) && __has_builtin(__builtin_amdgcn_s_wait_tensorcnt)
#define HAVE_TDM 1
#endif
#if __has_builtin(__builtin_amdgcn_s_cluster_barrier)
#define HAVE_CLUSTER_BAR 1
#endif
#if __has_builtin(__builtin_amdgcn_cluster_id_x)
#define HAVE_CLUSTER_ID 1
#endif
#endif

__launch_bounds__(32)
__global__ void tdm_probe(float* __restrict__ scratch)
{
#ifdef HAVE_TDM
    u32x4 g0 = { 0u, 0u, 0u, 0u };          // count=0: NULL tensor, no data moved
    i32x8 g1 = { 0, 0, 0, 0, 0, 0, 0, 0 };
    i32x4 g2 = { 0, 0, 0, 0 };
    i32x4 g3 = { 0, 0, 0, 0 };
    i32x8 g4 = { 0, 0, 0, 0, 0, 0, 0, 0 };
    __builtin_amdgcn_tensor_load_to_lds(g0, g1, g2, g3, g4, 0);
    __builtin_amdgcn_s_wait_tensorcnt(0);
#endif
#ifdef HAVE_CLUSTER_BAR
    __builtin_amdgcn_s_cluster_barrier();   // NOP when not dispatched in a cluster
#endif
    int cid = 0;
#ifdef HAVE_CLUSTER_ID
    cid = __builtin_amdgcn_cluster_id_x();  // 0 outside a cluster (deterministic)
#endif
    if (threadIdx.x == 0) { scratch[0] = 0.f; scratch[1] = (float)cid; }
}

// ---------------- host-side encoder layer ----------------
static void encoder_layer(hipStream_t stream,
                          const float* x_in, float* x_out,
                          const float* Wqkv, const float* bqkv,
                          const float* Wo,   const float* bo,
                          const float* W1,   const float* b1,
                          const float* W2,   const float* b2,
                          const float* g1,   const float* be1,
                          const float* g2,   const float* be2,
                          float* qkv, float* scores, float* attnout,
                          float* tmp, float* x1, float* ff)
{
    const int M = B_ * S_;
    const float RSQRT_HD = 0.08838834764831845f;   // 1/sqrt(128)
    dim3 blk(128);

    gemm_wmma<<<dim3(3 * D_ / 64, M / 64, 1), blk, 0, stream>>>(
        x_in, (long)D_, 0L, 0L, Wqkv, (long)D_, 0L, 0L, bqkv,
        qkv, (long)(3 * D_), 0L, 0L, M, 3 * D_, D_, 1, 1.0f, 1, 0);

    gemm_wmma<<<dim3(S_ / 64, S_ / 64, B_ * NHEAD_), blk, 0, stream>>>(
        qkv,      (long)(3 * D_), (long)S_ * 3 * D_, (long)HD_,
        qkv + D_, (long)(3 * D_), (long)S_ * 3 * D_, (long)HD_,
        nullptr,
        scores,   (long)S_, (long)NHEAD_ * S_ * S_, (long)S_ * S_,
        S_, S_, HD_, NHEAD_, RSQRT_HD, 1, 0);

    softmax_rows<<<dim3(B_ * NHEAD_ * S_ / 8), dim3(256), 0, stream>>>(scores);

    gemm_wmma<<<dim3(HD_ / 64, S_ / 64, B_ * NHEAD_), blk, 0, stream>>>(
        scores,       (long)S_, (long)NHEAD_ * S_ * S_, (long)S_ * S_,
        qkv + 2 * D_, (long)(3 * D_), (long)S_ * 3 * D_, (long)HD_,
        nullptr,
        attnout,      (long)D_, (long)S_ * D_, (long)HD_,
        S_, HD_, S_, NHEAD_, 1.0f, 0, 0);

    gemm_wmma<<<dim3(D_ / 64, M / 64, 1), blk, 0, stream>>>(
        attnout, (long)D_, 0L, 0L, Wo, (long)D_, 0L, 0L, bo,
        tmp, (long)D_, 0L, 0L, M, D_, D_, 1, 1.0f, 1, 0);

    add_ln<<<dim3(M), dim3(256), 0, stream>>>(x_in, tmp, g1, be1, x1);

    gemm_wmma<<<dim3(NHID_ / 64, M / 64, 1), blk, 0, stream>>>(
        x1, (long)D_, 0L, 0L, W1, (long)D_, 0L, 0L, b1,
        ff, (long)NHID_, 0L, 0L, M, NHID_, D_, 1, 1.0f, 1, 1);

    gemm_wmma<<<dim3(D_ / 64, M / 64, 1), blk, 0, stream>>>(
        ff, (long)NHID_, 0L, 0L, W2, (long)NHID_, 0L, 0L, b2,
        tmp, (long)D_, 0L, 0L, M, D_, NHID_, 1, 1.0f, 1, 0);

    add_ln<<<dim3(M), dim3(256), 0, stream>>>(x1, tmp, g2, be2, x_out);
}

// ---------------- entry point ----------------
extern "C" void kernel_launch(void* const* d_in, const int* in_sizes, int n_in,
                              void* d_out, int out_size, void* d_ws, size_t ws_size,
                              hipStream_t stream)
{
    const int*   inputs  = (const int*)  d_in[0];
    const float* noise   = (const float*)d_in[2];
    const float* resp    = (const float*)d_in[3];
    const int*   topk    = (const int*)  d_in[4];
    const float* emb     = (const float*)d_in[5];
    const float* r_Wqkv  = (const float*)d_in[6];
    const float* r_bqkv  = (const float*)d_in[7];
    const float* r_Wo    = (const float*)d_in[8];
    const float* r_bo    = (const float*)d_in[9];
    const float* r_W1    = (const float*)d_in[10];
    const float* r_b1    = (const float*)d_in[11];
    const float* r_W2    = (const float*)d_in[12];
    const float* r_b2    = (const float*)d_in[13];
    const float* r_ln1g  = (const float*)d_in[14];
    const float* r_ln1b  = (const float*)d_in[15];
    const float* r_ln2g  = (const float*)d_in[16];
    const float* r_ln2b  = (const float*)d_in[17];
    const float* e_Wqkv  = (const float*)d_in[18];
    const float* e_bqkv  = (const float*)d_in[19];
    const float* e_Wo    = (const float*)d_in[20];
    const float* e_bo    = (const float*)d_in[21];
    const float* e_W1    = (const float*)d_in[22];
    const float* e_b1    = (const float*)d_in[23];
    const float* e_W2    = (const float*)d_in[24];
    const float* e_b2    = (const float*)d_in[25];
    const float* e_ln1g  = (const float*)d_in[26];
    const float* e_ln1b  = (const float*)d_in[27];
    const float* e_ln2g  = (const float*)d_in[28];
    const float* e_ln2b  = (const float*)d_in[29];
    const float* dec_W   = (const float*)d_in[30];
    const float* gate_W  = (const float*)d_in[31];
    const float* gate_b  = (const float*)d_in[32];
    float* out = (float*)d_out;
    (void)in_sizes; (void)n_in; (void)out_size; (void)ws_size;

    float* ws = (float*)d_ws;
    size_t off = 0;
    auto alloc = [&](size_t n) { float* p = ws + off; off += n; return p; };
    const int M = B_ * S_;
    float* qkv     = alloc((size_t)M * 3 * D_);
    float* scores  = alloc((size_t)B_ * NHEAD_ * S_ * S_);
    float* attnout = alloc((size_t)M * D_);
    float* tmp     = alloc((size_t)M * D_);
    float* x1      = alloc((size_t)M * D_);
    float* ff      = alloc((size_t)M * NHID_);
    float* h0      = alloc((size_t)M * D_);
    float* rc      = alloc((size_t)M * D_);
    float* e0      = alloc((size_t)M * D_);
    float* e1      = alloc((size_t)M * D_);
    float* rw      = alloc((size_t)B_ * NUIDS_);
    float* rowloss = alloc((size_t)M);
    float* probes  = alloc(64);

    tdm_probe<<<dim3(1), dim3(32), 0, stream>>>(probes);

    // ---------- routing path ----------
    gather_emb<<<dim3(M), dim3(256), 0, stream>>>(inputs, emb, h0);
    encoder_layer(stream, h0, rc, r_Wqkv, r_bqkv, r_Wo, r_bo, r_W1, r_b1, r_W2, r_b2,
                  r_ln1g, r_ln1b, r_ln2g, r_ln2b, qkv, scores, attnout, tmp, x1, ff);
    gemm_wmma<<<dim3((NUIDS_ + 63) / 64, 1, 1), dim3(128), 0, stream>>>(
        rc + (size_t)(S_ - 1) * D_, (long)S_ * D_, 0L, 0L,
        gate_W, (long)D_, 0L, 0L, gate_b,
        rw, (long)NUIDS_, 0L, 0L, B_, NUIDS_, D_, 1, 32.0f /* sqrt(1024) */, 1, 0);
    gate_topk<<<dim3(1), dim3(256), 0, stream>>>(rw, noise, topk, out);

    // ---------- main path ----------
    encoder_layer(stream, resp, e0,
                  e_Wqkv + 0 * (size_t)3 * D_ * D_, e_bqkv + 0 * (size_t)3 * D_,
                  e_Wo   + 0 * (size_t)D_ * D_,     e_bo   + 0 * (size_t)D_,
                  e_W1   + 0 * (size_t)NHID_ * D_,  e_b1   + 0 * (size_t)NHID_,
                  e_W2   + 0 * (size_t)D_ * NHID_,  e_b2   + 0 * (size_t)D_,
                  e_ln1g + 0 * (size_t)D_, e_ln1b + 0 * (size_t)D_,
                  e_ln2g + 0 * (size_t)D_, e_ln2b + 0 * (size_t)D_,
                  qkv, scores, attnout, tmp, x1, ff);
    encoder_layer(stream, e0, e1,
                  e_Wqkv + 1 * (size_t)3 * D_ * D_, e_bqkv + 1 * (size_t)3 * D_,
                  e_Wo   + 1 * (size_t)D_ * D_,     e_bo   + 1 * (size_t)D_,
                  e_W1   + 1 * (size_t)NHID_ * D_,  e_b1   + 1 * (size_t)NHID_,
                  e_W2   + 1 * (size_t)D_ * NHID_,  e_b2   + 1 * (size_t)D_,
                  e_ln1g + 1 * (size_t)D_, e_ln1b + 1 * (size_t)D_,
                  e_ln2g + 1 * (size_t)D_, e_ln2b + 1 * (size_t)D_,
                  qkv, scores, attnout, tmp, x1, ff);

    ce_kernel<<<dim3(M / 16), dim3(128), 0, stream>>>(e1, dec_W, inputs, rowloss);
    loss_reduce<<<dim3(1), dim3(256), 0, stream>>>(rowloss, out);
}
